// GraphFingerprintsModelFC_18726057410985
// MI455X (gfx1250) — matrix operand order repaired and checked by reading, not compiled, not run on hardware
//
#include <hip/hip_runtime.h>

// ---------------- problem constants ----------------
#define NNODE 102400
#define NEDGE 204800
#define NGRAPH 2048
#define GDIM 256
#define NFEAT 39
#define EFEAT 10
#define FP_RAW 2192
#define FP_PAD 2208
#define CHUNK 25600   // GRU row chunk (mult of 128)

typedef __attribute__((ext_vector_type(16))) _Float16 v16h;
typedef __attribute__((ext_vector_type(8)))  float    v8f;

template <int ACT>
__device__ __forceinline__ float actf(float v) {
  if constexpr (ACT == 1) return fmaxf(v, 0.f);
  if constexpr (ACT == 2) return v > 0.f ? v : 0.01f * v;          // leaky relu 0.01
  if constexpr (ACT == 3) return v > 0.f ? v : (__expf(v) - 1.f);  // elu
  return v;
}

// ---------------- f32 -> f16 convert with padding + fused activation ----------------
// dst[r*ldDst + c] for c in [0,Cwrite); value = act(src[r*Csrc+c]) if r<Rsrc && c<Csrc else 0
template <int ACT>
__global__ void cvt_f16_kernel(const float* __restrict__ src, _Float16* __restrict__ dst,
                               int Rdst, int Rsrc, int Csrc, int Cwrite, int ldDst) {
  long idx = (long)blockIdx.x * 256 + threadIdx.x;
  long total = (long)Rdst * Cwrite;
  if (idx >= total) return;
  int r = (int)(idx / Cwrite);
  int c = (int)(idx % Cwrite);
  float v = (r < Rsrc && c < Csrc) ? actf<ACT>(src[(size_t)r * Csrc + c]) : 0.f;
  dst[(size_t)r * ldDst + c] = (_Float16)v;
}

// concat(node_feats[src[e]], edge_feats[e]) -> f16 [E,64] zero padded
__global__ void edge_concat_f16_kernel(const float* __restrict__ nf, const float* __restrict__ ef,
                                       const int* __restrict__ srcIdx, _Float16* __restrict__ dst) {
  long idx = (long)blockIdx.x * 256 + threadIdx.x;
  if (idx >= (long)NEDGE * 64) return;
  int e = (int)(idx >> 6);
  int c = (int)(idx & 63);
  float v = 0.f;
  if (c < NFEAT)              v = nf[(size_t)srcIdx[e] * NFEAT + c];
  else if (c < NFEAT + EFEAT) v = ef[(size_t)e * EFEAT + (c - NFEAT)];
  dst[idx] = (_Float16)v;
}

__global__ void fill_kernel(float* __restrict__ p, float v, int n) {
  int i = blockIdx.x * 256 + threadIdx.x;
  if (i < n) p[i] = v;
}

// ---------------- WMMA GEMM: C[M,N] = act(A_f16[M,K] * B_f16[K,N] + bias) ----------------
// block tile 128(M) x 128(N), 8 waves (4 in M, 2 in N), wave tile 32x64 = 2x4 WMMA frags
// A frag (16x32, lane=M row, vgpr pairs=K per ISA 7.12.2) : 2x ds_load_b128 per frag
// B frag (32x16, lane=N col, vector elems = 16 K values per ISA 7.12.4): 16 strided 16-bit LDS reads
#define LDA_T 40
#define LDB_T 136
template <int ACT>
__global__ __launch_bounds__(256) void gemm_f16_wmma(
    const _Float16* __restrict__ A, int lda,
    const _Float16* __restrict__ Bw, int ldb,
    const float* __restrict__ bias,
    float* __restrict__ C, int ldc,
    int M, int N, int K) {
  __shared__ __align__(16) _Float16 sA[128 * LDA_T];
  __shared__ __align__(16) _Float16 sB[32 * LDB_T];
  int tid = threadIdx.x;
  int lane = tid & 31;
  int wave = tid >> 5;
  int wy = wave & 3;   // M direction (4 x 32)
  int wx = wave >> 2;  // N direction (2 x 64)
  int mBase = blockIdx.y * 128;
  int nBase = blockIdx.x * 128;

  v8f acc[2][4] = {};

  for (int kb = 0; kb < K; kb += 32) {
    // cooperative load A tile 128x32 (512 chunks of 8 f16, 2 per thread)
#pragma unroll
    for (int it = 0; it < 2; ++it) {
      int ch = tid + it * 256;
      int r = ch >> 2;
      int c8 = (ch & 3) * 8;
      const _Float16* gp = A + (size_t)(mBase + r) * lda + kb + c8;
      *reinterpret_cast<uint4*>(&sA[r * LDA_T + c8]) = *reinterpret_cast<const uint4*>(gp);
    }
    // cooperative load B tile 32x128 (512 chunks of 8 f16, 2 per thread)
#pragma unroll
    for (int it = 0; it < 2; ++it) {
      int ch = tid + it * 256;
      int r = ch >> 4;
      int c8 = (ch & 15) * 8;
      const _Float16* gp = Bw + (size_t)(kb + r) * ldb + nBase + c8;
      *reinterpret_cast<uint4*>(&sB[r * LDB_T + c8]) = *reinterpret_cast<const uint4*>(gp);
    }
    __syncthreads();

    int g = lane >> 4;
    int lm = lane & 15;
    v16h af[2], bf[4];
    // A fragments: lane = M row; vgpr j holds K pair (j<4: k=g*8+2j, j>=4: k=16+g*8+2(j-4))
#pragma unroll
    for (int mi = 0; mi < 2; ++mi) {
      int r = wy * 32 + mi * 16 + lm;
#pragma unroll
      for (int j = 0; j < 8; ++j) {
        int k = (j < 4) ? (g * 8 + 2 * j) : (16 + g * 8 + 2 * (j - 4));
        af[mi][2 * j]     = sA[r * LDA_T + k];
        af[mi][2 * j + 1] = sA[r * LDA_T + k + 1];
      }
    }
    // B fragments: lane = N col; elem e holds k (e<8: g*8+e, e>=8: 16+g*8+(e-8))
    {
#pragma unroll
      for (int ni = 0; ni < 4; ++ni) {
        const _Float16* bp = &sB[wx * 64 + ni * 16 + lm];
#pragma unroll
        for (int e = 0; e < 16; ++e) {
          int k = (e < 8) ? (g * 8 + e) : (16 + g * 8 + (e - 8));
          bf[ni][e] = bp[k * LDB_T];
        }
      }
    }
#pragma unroll
    for (int mi = 0; mi < 2; ++mi)
#pragma unroll
      for (int ni = 0; ni < 4; ++ni)
        acc[mi][ni] = __builtin_amdgcn_wmma_f32_16x16x32_f16(
            false, af[mi], false, bf[ni], (short)0, acc[mi][ni], false, false);
    __syncthreads();
  }

  int g = lane >> 4;
  int lm = lane & 15;
#pragma unroll
  for (int mi = 0; mi < 2; ++mi)
#pragma unroll
    for (int ni = 0; ni < 4; ++ni)
#pragma unroll
      for (int i = 0; i < 8; ++i) {
        int row = mBase + wy * 32 + mi * 16 + i + 8 * g;
        int col = nBase + wx * 64 + ni * 16 + lm;
        float v = acc[mi][ni][i] + bias[col];
        C[(size_t)row * ldc + col] = actf<ACT>(v);
      }
}

// ---------------- attention logit: lrelu( dot(concat(A[ra], B[rb]), w) + b ), one wave/row ----
__global__ __launch_bounds__(256) void dot512_kernel(
    const float* __restrict__ A, const int* __restrict__ idxA, int reluA,
    const float* __restrict__ Bsrc, const int* __restrict__ idxB,
    const float* __restrict__ w, const float* __restrict__ bias,
    float* __restrict__ out, int rows) {
  int row = blockIdx.x * 8 + (threadIdx.x >> 5);
  if (row >= rows) return;
  int lane = threadIdx.x & 31;
  int ra = idxA ? idxA[row] : row;
  int rb = idxB ? idxB[row] : row;
  const float* ap = A + (size_t)ra * GDIM;
  const float* bp = Bsrc + (size_t)rb * GDIM;
  float s = 0.f;
#pragma unroll 4
  for (int t = lane; t < GDIM; t += 32) {
    float av = ap[t];
    if (reluA) av = fmaxf(av, 0.f);
    s += av * w[t];
    s += bp[t] * w[GDIM + t];
  }
  for (int off = 16; off > 0; off >>= 1) s += __shfl_xor(s, off, 32);
  if (lane == 0) {
    float v = s + bias[0];
    out[row] = v > 0.f ? v : 0.01f * v;
  }
}

// ---------------- segment softmax helpers ----------------
__device__ __forceinline__ void atomicMaxFloat(float* addr, float val) {
  unsigned int* ua = (unsigned int*)addr;
  unsigned int old = *ua;
  while (__uint_as_float(old) < val) {
    unsigned int assumed = old;
    old = atomicCAS(ua, assumed, __float_as_uint(val));
    if (old == assumed) break;
  }
}
__global__ void seg_max_kernel(const float* __restrict__ logits, const int* __restrict__ seg,
                               float* __restrict__ segMax, int n) {
  int i = blockIdx.x * 256 + threadIdx.x;
  if (i < n) atomicMaxFloat(&segMax[seg[i]], logits[i]);
}
__global__ void seg_expsum_kernel(const float* __restrict__ logits, const int* __restrict__ seg,
                                  const float* __restrict__ segMax, float* __restrict__ segSum,
                                  float* __restrict__ e, int n) {
  int i = blockIdx.x * 256 + threadIdx.x;
  if (i >= n) return;
  float v = __expf(logits[i] - segMax[seg[i]]);
  e[i] = v;
  atomicAdd(&segSum[seg[i]], v);
}
// out[seg[r]] += (e[r]/segSum[seg[r]]) * V[idxV?idxV[r]:r]  (one block of 256 per row)
__global__ __launch_bounds__(256) void attn_scatter_kernel(
    const float* __restrict__ e, const float* __restrict__ segSum, const int* __restrict__ seg,
    const float* __restrict__ V, const int* __restrict__ idxV, float* __restrict__ out) {
  int r = blockIdx.x;
  int f = threadIdx.x;
  float a = e[r] / segSum[seg[r]];
  int vr = idxV ? idxV[r] : r;
  atomicAdd(&out[(size_t)seg[r] * GDIM + f], a * V[(size_t)vr * GDIM + f]);
}
__global__ __launch_bounds__(256) void seg_sum_kernel(const float* __restrict__ X,
                                                      const int* __restrict__ seg,
                                                      float* __restrict__ out) {
  int r = blockIdx.x;
  int f = threadIdx.x;
  atomicAdd(&out[(size_t)seg[r] * GDIM + f], X[(size_t)r * GDIM + f]);
}

// ---------------- GRU elementwise (torch GRUCell) + relu ----------------
__global__ __launch_bounds__(256) void gru_kernel(const float* __restrict__ gi,
                                                  const float* __restrict__ gh,
                                                  const float* __restrict__ h,
                                                  float* __restrict__ out) {
  int r = blockIdx.x;
  int f = threadIdx.x;
  const float* gip = gi + (size_t)r * 768;
  const float* ghp = gh + (size_t)r * 768;
  float ir = gip[f], iz = gip[256 + f], in = gip[512 + f];
  float hr = ghp[f], hz = ghp[256 + f], hn = ghp[512 + f];
  float rg = 1.f / (1.f + __expf(-(ir + hr)));
  float zg = 1.f / (1.f + __expf(-(iz + hz)));
  float ng = tanhf(in + rg * hn);
  float hv = (1.f - zg) * ng + zg * h[(size_t)r * GDIM + f];
  out[(size_t)r * GDIM + f] = fmaxf(hv, 0.f);
}

// ---------------- final head: out[r] = dot(h[r,:128], w) + b ----------------
__global__ void head_kernel(const float* __restrict__ h, const float* __restrict__ w,
                            const float* __restrict__ b, float* __restrict__ out, int rows) {
  int r = blockIdx.x * 256 + threadIdx.x;
  if (r >= rows) return;
  float s = b[0];
#pragma unroll 8
  for (int k = 0; k < 128; ++k) s += h[(size_t)r * 128 + k] * w[k];
  out[r] = s;
}

// =====================================================================================
extern "C" void kernel_launch(void* const* d_in, const int* in_sizes, int n_in,
                              void* d_out, int out_size, void* d_ws, size_t ws_size,
                              hipStream_t stream) {
  (void)in_sizes; (void)n_in; (void)out_size; (void)ws_size;
  // ---- input map (dict insertion order, params flattened depth-first) ----
  const float* node_feats   = (const float*)d_in[0];
  const float* edge_feats   = (const float*)d_in[1];
  const float* fingerprints = (const float*)d_in[2];
  auto PF = [&](int i) -> const float* { return (const float*)d_in[i]; };
  const int* srcIdx    = (const int*)d_in[47];
  const int* dstIdx    = (const int*)d_in[48];
  const int* nodeGraph = (const int*)d_in[49];
  float* out = (float*)d_out;

  // ---- workspace bump allocator ----
  size_t off = 0;
  auto alloc = [&](size_t bytes) -> char* {
    off = (off + 255) & ~(size_t)255;
    char* p = (char*)d_ws + off;
    off += bytes;
    return p;
  };
  auto af16 = [&](size_t n) { return (_Float16*)alloc(n * sizeof(_Float16)); };
  auto af32 = [&](size_t n) { return (float*)alloc(n * sizeof(float)); };

  // f16 weights
  _Float16* w_pn16  = af16(64 * 256);
  _Float16* w_pe116 = af16(64 * 256);
  _Float16* w_et16  = af16(256 * 256);
  _Float16* w_g0ih  = af16(256 * 768);
  _Float16* w_g0hh  = af16(256 * 768);
  _Float16* w_l1pn  = af16(256 * 256);
  _Float16* w_g1ih  = af16(256 * 768);
  _Float16* w_g1hh  = af16(256 * 768);
  _Float16* w_rpn[2]; _Float16* w_rih[2]; _Float16* w_rhh[2];
  for (int t = 0; t < 2; ++t) {
    w_rpn[t] = af16(256 * 256);
    w_rih[t] = af16(256 * 768);
    w_rhh[t] = af16(256 * 768);
  }
  _Float16* w_fc1 = af16((size_t)FP_PAD * 256);
  _Float16* w_fc2 = af16(256 * 256);
  _Float16* w_p1  = af16(512 * 128);
  // f16 activations
  _Float16* nfeat16 = af16((size_t)NNODE * 64);
  _Float16* econ16  = af16((size_t)NEDGE * 64);
  _Float16* he1_16  = af16((size_t)NEDGE * 256);
  _Float16* X16     = af16((size_t)NNODE * 256);
  _Float16* H16     = af16((size_t)NNODE * 256);
  _Float16* fp16b   = af16((size_t)NGRAPH * FP_PAD);
  _Float16* xb16    = af16((size_t)NGRAPH * 256);
  _Float16* gb16    = af16((size_t)NGRAPH * 256);
  _Float16* comb16  = af16((size_t)NGRAPH * 512);
  // f32 buffers
  float* hvnew  = af32((size_t)NNODE * 256);
  float* he1    = af32((size_t)NEDGE * 256);
  float* etout  = af32((size_t)NEDGE * 256);
  float* ctx    = af32((size_t)NNODE * 256);
  float* hvA    = af32((size_t)NNODE * 256);
  float* hvB    = af32((size_t)NNODE * 256);
  float* pnout  = af32((size_t)NNODE * 256);
  float* gi_c   = af32((size_t)CHUNK * 768);
  float* gh_c   = af32((size_t)CHUNK * 768);
  float* logitsE = af32(NEDGE);
  float* segMaxN = af32(NNODE);
  float* segSumN = af32(NNODE);
  float* segMaxB = af32(NGRAPH);
  float* segSumB = af32(NGRAPH);
  float* gfeatA = af32((size_t)NGRAPH * 256);
  float* gfeatB = af32((size_t)NGRAPH * 256);
  float* grepr  = af32((size_t)NGRAPH * 256);
  float* giB    = af32((size_t)NGRAPH * 768);
  float* ghB    = af32((size_t)NGRAPH * 768);
  float* fp1    = af32((size_t)NGRAPH * 256);
  float* fpout  = af32((size_t)NGRAPH * 256);
  float* p1out  = af32((size_t)NGRAPH * 128);

  // ---- helpers ----
  auto cvt = [&](const float* s, _Float16* d, int Rdst, int Rsrc, int Csrc, int Cwrite,
                 int ldDst, int act) {
    long total = (long)Rdst * Cwrite;
    int grid = (int)((total + 255) / 256);
    if (act == 3)
      cvt_f16_kernel<3><<<grid, 256, 0, stream>>>(s, d, Rdst, Rsrc, Csrc, Cwrite, ldDst);
    else
      cvt_f16_kernel<0><<<grid, 256, 0, stream>>>(s, d, Rdst, Rsrc, Csrc, Cwrite, ldDst);
  };
  auto gemm = [&](const _Float16* A, int lda, const _Float16* Bw, int ldb, const float* bias,
                  float* C, int ldc, int M, int N, int K, int act) {
    dim3 grid(N / 128, M / 128);
    if (act == 1)
      gemm_f16_wmma<1><<<grid, 256, 0, stream>>>(A, lda, Bw, ldb, bias, C, ldc, M, N, K);
    else if (act == 2)
      gemm_f16_wmma<2><<<grid, 256, 0, stream>>>(A, lda, Bw, ldb, bias, C, ldc, M, N, K);
    else
      gemm_f16_wmma<0><<<grid, 256, 0, stream>>>(A, lda, Bw, ldb, bias, C, ldc, M, N, K);
  };
  auto fillf = [&](float* p, float v, int n) {
    fill_kernel<<<(n + 255) / 256, 256, 0, stream>>>(p, v, n);
  };
  auto softmax_edges = [&](float* logits, const int* seg, float* smax, float* ssum, int rows,
                           int nseg) {
    fillf(smax, -3e38f, nseg);
    fillf(ssum, 0.f, nseg);
    seg_max_kernel<<<(rows + 255) / 256, 256, 0, stream>>>(logits, seg, smax, rows);
    seg_expsum_kernel<<<(rows + 255) / 256, 256, 0, stream>>>(logits, seg, smax, ssum, logits, rows);
  };

  // ---- convert weights ----
  cvt(PF(3),  w_pn16,  64, NFEAT, 256, 256, 256, 0);
  cvt(PF(5),  w_pe116, 64, NFEAT + EFEAT, 256, 256, 256, 0);
  cvt(PF(9),  w_et16,  256, 256, 256, 256, 256, 0);
  cvt(PF(11), w_g0ih,  256, 256, 768, 768, 768, 0);
  cvt(PF(13), w_g0hh,  256, 256, 768, 768, 768, 0);
  cvt(PF(17), w_l1pn,  256, 256, 256, 256, 256, 0);
  cvt(PF(19), w_g1ih,  256, 256, 768, 768, 768, 0);
  cvt(PF(21), w_g1hh,  256, 256, 768, 768, 768, 0);
  for (int t = 0; t < 2; ++t) {
    int b = 23 + 8 * t;
    cvt(PF(b + 2), w_rpn[t], 256, 256, 256, 256, 256, 0);
    cvt(PF(b + 4), w_rih[t], 256, 256, 768, 768, 768, 0);
    cvt(PF(b + 6), w_rhh[t], 256, 256, 768, 768, 768, 0);
  }
  cvt(PF(39), w_fc1, FP_PAD, FP_RAW, 256, 256, 256, 0);
  cvt(PF(41), w_fc2, 256, 256, 256, 256, 256, 0);
  cvt(PF(43), w_p1,  512, 512, 128, 128, 128, 0);

  // ---- GetContext layer ----
  cvt(node_feats, nfeat16, NNODE, NNODE, NFEAT, 64, 64, 0);
  gemm(nfeat16, 64, w_pn16, 256, PF(4), hvnew, 256, NNODE, 256, 64, 2);  // hv_new = lrelu
  edge_concat_f16_kernel<<<(NEDGE * 64 + 255) / 256, 256, 0, stream>>>(node_feats, edge_feats,
                                                                       srcIdx, econ16);
  gemm(econ16, 64, w_pe116, 256, PF(6), he1, 256, NEDGE, 256, 64, 2);    // he1 = lrelu
  cvt(he1, he1_16, NEDGE, NEDGE, 256, 256, 256, 0);
  gemm(he1_16, 256, w_et16, 256, PF(10), etout, 256, NEDGE, 256, 256, 0);
  dot512_kernel<<<(NEDGE + 7) / 8, 256, 0, stream>>>(hvnew, dstIdx, 0, he1, nullptr, PF(7),
                                                     PF(8), logitsE, NEDGE);
  softmax_edges(logitsE, dstIdx, segMaxN, segSumN, NEDGE, NNODE);
  fillf(ctx, 0.f, NNODE * 256);
  attn_scatter_kernel<<<NEDGE, 256, 0, stream>>>(logitsE, segSumN, dstIdx, etout, nullptr, ctx);
  cvt(ctx, X16, NNODE, NNODE, 256, 256, 256, 3);    // elu(c)
  cvt(hvnew, H16, NNODE, NNODE, 256, 256, 256, 0);  // h
  for (int c = 0; c < NNODE / CHUNK; ++c) {
    size_t ro = (size_t)c * CHUNK;
    gemm(X16 + ro * 256, 256, w_g0ih, 768, PF(12), gi_c, 768, CHUNK, 768, 256, 0);
    gemm(H16 + ro * 256, 256, w_g0hh, 768, PF(14), gh_c, 768, CHUNK, 768, 256, 0);
    gru_kernel<<<CHUNK, 256, 0, stream>>>(gi_c, gh_c, hvnew + ro * 256, hvA + ro * 256);
  }

  // ---- GNN layer 2 (AttentiveGRU2) ----
  dot512_kernel<<<(NEDGE + 7) / 8, 256, 0, stream>>>(hvA, dstIdx, 0, hvA, srcIdx, PF(15),
                                                     PF(16), logitsE, NEDGE);
  softmax_edges(logitsE, dstIdx, segMaxN, segSumN, NEDGE, NNODE);
  cvt(hvA, H16, NNODE, NNODE, 256, 256, 256, 0);
  gemm(H16, 256, w_l1pn, 256, PF(18), pnout, 256, NNODE, 256, 256, 0);
  fillf(ctx, 0.f, NNODE * 256);
  attn_scatter_kernel<<<NEDGE, 256, 0, stream>>>(logitsE, segSumN, dstIdx, pnout, srcIdx, ctx);
  cvt(ctx, X16, NNODE, NNODE, 256, 256, 256, 3);
  for (int c = 0; c < NNODE / CHUNK; ++c) {
    size_t ro = (size_t)c * CHUNK;
    gemm(X16 + ro * 256, 256, w_g1ih, 768, PF(20), gi_c, 768, CHUNK, 768, 256, 0);
    gemm(H16 + ro * 256, 256, w_g1hh, 768, PF(22), gh_c, 768, CHUNK, 768, 256, 0);
    gru_kernel<<<CHUNK, 256, 0, stream>>>(gi_c, gh_c, hvA + ro * 256, hvB + ro * 256);
  }

  // ---- readout ----
  fillf(gfeatA, 0.f, NGRAPH * 256);
  seg_sum_kernel<<<NNODE, 256, 0, stream>>>(hvB, nodeGraph, gfeatA);
  cvt(hvB, H16, NNODE, NNODE, 256, 256, 256, 0);
  float* gcur = gfeatA;
  float* gnext = gfeatB;
  for (int t = 0; t < 2; ++t) {
    int b = 23 + 8 * t;
    gemm(H16, 256, w_rpn[t], 256, PF(b + 3), pnout, 256, NNODE, 256, 256, 0);
    dot512_kernel<<<(NNODE + 7) / 8, 256, 0, stream>>>(gcur, nodeGraph, 1, hvB, nullptr,
                                                       PF(b + 0), PF(b + 1), logitsE, NNODE);
    softmax_edges(logitsE, nodeGraph, segMaxB, segSumB, NNODE, NGRAPH);
    fillf(grepr, 0.f, NGRAPH * 256);
    attn_scatter_kernel<<<NNODE, 256, 0, stream>>>(logitsE, segSumB, nodeGraph, pnout, nullptr,
                                                   grepr);
    cvt(grepr, xb16, NGRAPH, NGRAPH, 256, 256, 256, 3);  // elu
    cvt(gcur, gb16, NGRAPH, NGRAPH, 256, 256, 256, 0);
    gemm(xb16, 256, w_rih[t], 768, PF(b + 5), giB, 768, NGRAPH, 768, 256, 0);
    gemm(gb16, 256, w_rhh[t], 768, PF(b + 7), ghB, 768, NGRAPH, 768, 256, 0);
    gru_kernel<<<NGRAPH, 256, 0, stream>>>(giB, ghB, gcur, gnext);
    float* tmp = gcur; gcur = gnext; gnext = tmp;
  }

  // ---- fingerprint FC + predict head ----
  cvt(fingerprints, fp16b, NGRAPH, NGRAPH, FP_RAW, FP_PAD, FP_PAD, 0);
  gemm(fp16b, FP_PAD, w_fc1, 256, PF(40), fp1, 256, NGRAPH, 256, FP_PAD, 1);  // relu
  cvt(fp1, xb16, NGRAPH, NGRAPH, 256, 256, 256, 0);
  gemm(xb16, 256, w_fc2, 256, PF(42), fpout, 256, NGRAPH, 256, 256, 0);
  cvt(gcur, comb16, NGRAPH, NGRAPH, 256, 256, 512, 0);         // comb[:, :256]
  cvt(fpout, comb16 + 256, NGRAPH, NGRAPH, 256, 256, 512, 0);  // comb[:, 256:]
  gemm(comb16, 512, w_p1, 128, PF(44), p1out, 128, NGRAPH, 128, 512, 1);  // relu
  head_kernel<<<(NGRAPH + 255) / 256, 256, 0, stream>>>(p1out, PF(45), PF(46), out, NGRAPH);
}
